// DyGCN_7069516169749
// MI455X (gfx1250) — matrix operand order repaired and checked by reading, not compiled, not run on hardware
//
#include <hip/hip_runtime.h>
#include <hip/hip_bf16.h>

// ---------------------------------------------------------------------------
// CDNA5 (gfx1250) implementation of the DyGCN step.
// All GEMMs run on v_wmma_f32_16x16x32_f16 (f32 HBM -> f16 LDS -> f32 acc).
// ---------------------------------------------------------------------------

typedef _Float16 h16v __attribute__((ext_vector_type(16)));
typedef _Float16 h8v  __attribute__((ext_vector_type(8)));
typedef _Float16 h4v  __attribute__((ext_vector_type(4)));
typedef float    f8v  __attribute__((ext_vector_type(8)));
typedef float    f4v  __attribute__((ext_vector_type(4)));

#define LDP 136   // padded LDS row stride (halfs): 136*2B = 272B = 17*16B -> aligned frag reads

// ---- WMMA fragment loads from LDS (ISA 7.12.2 layouts) --------------------
__device__ __forceinline__ h16v load_afrag(const _Float16* As, int m16, int k0, int lane) {
    int m  = m16 + (lane & 15);
    int kb = (lane >> 4) << 3;                    // 0 or 8
    const _Float16* p = As + m * LDP + k0 + kb;
    h8v lo = *(const h8v*)(p);
    h8v hi = *(const h8v*)(p + 16);
    h16v a;
#pragma unroll
    for (int i = 0; i < 8; ++i) { a[i] = lo[i]; a[8 + i] = hi[i]; }
    return a;
}

__device__ __forceinline__ h16v load_bfrag(const _Float16* Bt, int n16, int k0, int lane) {
    int n  = n16 + (lane & 15);
    int kb = (lane >> 4) << 3;                    // 0 or 8
    const _Float16* p = Bt + n * LDP + k0 + kb;
    h8v lo = *(const h8v*)(p);
    h8v hi = *(const h8v*)(p + 16);
    h16v b;
#pragma unroll
    for (int i = 0; i < 8; ++i) { b[i] = lo[i]; b[8 + i] = hi[i]; }
    return b;
}

// ---------------------------------------------------------------------------
// Generic GEMM: C[M,128] (+)= act_a(A[M,128]) @ B[:,col0:col0+128]
// a_relu: apply relu to A during f32->f16 staging (fuses inter-layer relu).
// act: 0=none 1=relu 2=sigmoid.  bias pre-offset by caller (indexed by n).
// ---------------------------------------------------------------------------
__global__ __launch_bounds__(256) void gemm128_kernel(
    const float* __restrict__ A, int M, int a_relu,
    const float* __restrict__ B, int ldb, int col0,
    const float* __restrict__ bias,
    float* __restrict__ C, int acc, int act)
{
    __shared__ _Float16 As[64 * LDP];
    __shared__ _Float16 Bt[128 * LDP];
    const int tid  = threadIdx.x;
    const int row0 = blockIdx.x * 64;

    // stage A (vec4 loads, optional fused relu)
    for (int i4 = tid; i4 < 64 * 32; i4 += 256) {
        int r = i4 >> 5, k4 = (i4 & 31) << 2;
        int gr = row0 + r; if (gr >= M) gr = M - 1;
        f4v v = *(const f4v*)(A + gr * 128 + k4);
        h4v h;
#pragma unroll
        for (int j = 0; j < 4; ++j) {
            float w = v[j];
            if (a_relu) w = w > 0.0f ? w : 0.0f;
            h[j] = (_Float16)w;
        }
        *(h4v*)(As + r * LDP + k4) = h;
    }
    // stage B transposed (vec4 coalesced reads, 16-bit scatter writes)
    for (int i4 = tid; i4 < 128 * 32; i4 += 256) {
        int k = i4 >> 5, n4 = (i4 & 31) << 2;
        f4v v = *(const f4v*)(B + k * ldb + col0 + n4);
#pragma unroll
        for (int j = 0; j < 4; ++j) Bt[(n4 + j) * LDP + k] = (_Float16)v[j];
    }
    __syncthreads();

    const int wave = tid >> 5, lane = tid & 31;
    const int wm = wave >> 1, wn = wave & 1;
    const int m16 = wm * 16;
    const int nlo = lane & 15, mb = (lane >> 4) * 8;

    f8v cr[4];
#pragma unroll
    for (int t = 0; t < 4; ++t) {
        int n = wn * 64 + t * 16 + nlo;
#pragma unroll
        for (int i = 0; i < 8; ++i) {
            int gr = row0 + m16 + mb + i;
            cr[t][i] = (acc && gr < M) ? C[gr * 128 + n] : 0.0f;
        }
    }
#pragma unroll
    for (int kk = 0; kk < 4; ++kk) {
        int k0 = kk * 32;
        h16v a = load_afrag(As, m16, k0, lane);
#pragma unroll
        for (int t = 0; t < 4; ++t) {
            h16v b = load_bfrag(Bt, wn * 64 + t * 16, k0, lane);
            cr[t] = __builtin_amdgcn_wmma_f32_16x16x32_f16(
                        false, a, false, b, (short)0, cr[t], false, false);
        }
    }
#pragma unroll
    for (int t = 0; t < 4; ++t) {
        int n = wn * 64 + t * 16 + nlo;
        float bv = bias ? bias[n] : 0.0f;
#pragma unroll
        for (int i = 0; i < 8; ++i) {
            int gr = row0 + m16 + mb + i;
            if (gr < M) {
                float v = cr[t][i] + bv;
                if (act == 1)      v = v > 0.0f ? v : 0.0f;
                else if (act == 2) v = 1.0f / (1.0f + __expf(-v));
                C[gr * 128 + n] = v;
            }
        }
    }
}

// ---------------------------------------------------------------------------
// Fused edge-sampling kernel: 64 edges/block.
//   pre = prev[src]@Ws1[0:128] + prev[dst]@Ws1[128:256] + te@Ws1[256:384] + b1
//   z   = sigmoid(relu(pre) @ Ws2 + b2)
// ---------------------------------------------------------------------------
union SampleSmem {
    struct { _Float16 As[64 * LDP]; _Float16 Bt[128 * LDP]; } s;
    float P[64 * 132];
};

__global__ __launch_bounds__(256) void sample_kernel(
    const float* __restrict__ prev,
    const int* __restrict__ src, const int* __restrict__ dst,
    const float* __restrict__ etime, const unsigned* __restrict__ tminbits,
    const float* __restrict__ w_time, const float* __restrict__ b_time,
    const float* __restrict__ w_s1, const float* __restrict__ b_s1,
    const float* __restrict__ w_s2, const float* __restrict__ b_s2,
    float* __restrict__ z, int E)
{
    __shared__ SampleSmem u;
    __shared__ int   es[64], ed[64];
    __shared__ float tl[64];

    const int tid = threadIdx.x;
    const int e0  = blockIdx.x * 64;
    if (tid < 64) {
        int e = e0 + tid; if (e >= E) e = E - 1;
        es[tid] = src[e];
        ed[tid] = dst[e];
        tl[tid] = etime[e] - __uint_as_float(*tminbits) + 1.0f;
    }
    __syncthreads();

    const int wave = tid >> 5, lane = tid & 31;
    const int wm = wave >> 1, wn = wave & 1;
    const int m16 = wm * 16;
    const int nlo = lane & 15, mb = (lane >> 4) * 8;

    f8v cr[4];
#pragma unroll
    for (int t = 0; t < 4; ++t)
#pragma unroll
        for (int i = 0; i < 8; ++i) cr[t][i] = 0.0f;

    for (int p = 0; p < 3; ++p) {
        for (int i4 = tid; i4 < 64 * 32; i4 += 256) {
            int r = i4 >> 5, k4 = (i4 & 31) << 2;
            f4v v;
            if (p == 0)      v = *(const f4v*)(prev + es[r] * 128 + k4);
            else if (p == 1) v = *(const f4v*)(prev + ed[r] * 128 + k4);
            else {
                f4v wt = *(const f4v*)(w_time + k4);
                f4v bt = *(const f4v*)(b_time + k4);
                float t = tl[r];
#pragma unroll
                for (int j = 0; j < 4; ++j) v[j] = __cosf(t * wt[j] + bt[j]);
            }
            h4v h;
#pragma unroll
            for (int j = 0; j < 4; ++j) h[j] = (_Float16)v[j];
            *(h4v*)(u.s.As + r * LDP + k4) = h;
        }
        for (int i4 = tid; i4 < 128 * 32; i4 += 256) {
            int k = i4 >> 5, n4 = (i4 & 31) << 2;
            f4v v = *(const f4v*)(w_s1 + (p * 128 + k) * 128 + n4);
#pragma unroll
            for (int j = 0; j < 4; ++j) u.s.Bt[(n4 + j) * LDP + k] = (_Float16)v[j];
        }
        __syncthreads();
#pragma unroll
        for (int kk = 0; kk < 4; ++kk) {
            int k0 = kk * 32;
            h16v a = load_afrag(u.s.As, m16, k0, lane);
#pragma unroll
            for (int t = 0; t < 4; ++t) {
                h16v b = load_bfrag(u.s.Bt, wn * 64 + t * 16, k0, lane);
                cr[t] = __builtin_amdgcn_wmma_f32_16x16x32_f16(
                            false, a, false, b, (short)0, cr[t], false, false);
            }
        }
        __syncthreads();
    }
    // relu(pre + b1) -> P
#pragma unroll
    for (int t = 0; t < 4; ++t) {
        int n = wn * 64 + t * 16 + nlo;
        float bv = b_s1[n];
#pragma unroll
        for (int i = 0; i < 8; ++i) {
            float v = cr[t][i] + bv;
            u.P[(m16 + mb + i) * 132 + n] = v > 0.0f ? v : 0.0f;
        }
    }
    __syncthreads();
    if (tid < 64) {
        int e = e0 + tid;
        if (e < E) {
            float s = 0.0f;
#pragma unroll 4
            for (int k = 0; k < 128; ++k) s += u.P[tid * 132 + k] * w_s2[k];
            s += b_s2[0];
            z[e] = 1.0f / (1.0f + __expf(-s));
        }
    }
}

// ---------------------------------------------------------------------------
// Small utility kernels
// ---------------------------------------------------------------------------
__global__ void init_kernel(unsigned* hist, unsigned* tminbits, float* deg, int N) {
    int i = blockIdx.x * blockDim.x + threadIdx.x;
    if (i < 4096) hist[i] = 0u;
    if (i == 0)   *tminbits = 0x7F800000u;   // +inf
    if (i < N)    deg[i] = 1.0f;             // self-loop weight
}

__global__ void tmin_kernel(const float* __restrict__ t, int n, unsigned* out) {
    int i = blockIdx.x * blockDim.x + threadIdx.x;
    unsigned v = 0x7F800000u;
    if (i < n) v = __float_as_uint(t[i]);    // t >= 0 -> bit order == float order
#pragma unroll
    for (int o = 16; o; o >>= 1) {
        unsigned w = __shfl_xor(v, o, 32);
        v = v < w ? v : w;
    }
    if ((threadIdx.x & 31) == 0) atomicMin(out, v);
}

__global__ void hist_kernel(const float* __restrict__ z, int E, unsigned* hist) {
    int e = blockIdx.x * blockDim.x + threadIdx.x;
    if (e < E) {
        int b = (int)(z[e] * 4096.0f);
        b = b < 0 ? 0 : (b > 4095 ? 4095 : b);
        atomicAdd(&hist[b], 1u);
    }
}

__global__ void thresh_kernel(const unsigned* __restrict__ hist, float* thresh, int K) {
    if (threadIdx.x == 0 && blockIdx.x == 0) {
        unsigned cnt = 0; int b = 4095;
        for (; b > 0; --b) { cnt += hist[b]; if (cnt >= (unsigned)K) break; }
        *thresh = (float)b * (1.0f / 4096.0f);
    }
}

__global__ void deg_kernel(const int* __restrict__ dst, const float* __restrict__ z,
                           const float* __restrict__ thresh, float* deg, int E) {
    int e = blockIdx.x * blockDim.x + threadIdx.x;
    if (e < E) {
        float zi = z[e];
        if (zi >= *thresh) atomicAdd(&deg[dst[e]], zi);
    }
}

__global__ void dinv_kernel(float* deg, int N) {
    int i = blockIdx.x * blockDim.x + threadIdx.x;
    if (i < N) { float d = deg[i]; deg[i] = d > 0.0f ? rsqrtf(d) : 0.0f; }
}

// agg = dinv^2 * m + bias   (self-loop term + bias init), vec4
__global__ void agg_init_kernel(const float* __restrict__ m, const float* __restrict__ dinv,
                                const float* __restrict__ bias, float* agg, int NH4) {
    int i4 = blockIdx.x * blockDim.x + threadIdx.x;
    if (i4 < NH4) {
        int row = i4 >> 5, c4 = (i4 & 31) << 2;
        float di = dinv[row];
        float d2 = di * di;
        f4v mv = *(const f4v*)(m + i4 * 4);
        f4v bv = *(const f4v*)(bias + c4);
        f4v r;
#pragma unroll
        for (int j = 0; j < 4; ++j) r[j] = d2 * mv[j] + bv[j];
        *(f4v*)(agg + i4 * 4) = r;
    }
}

// one wave per edge: agg[dst] += norm * m[src]   (b128 loads, 4 atomics/lane)
__global__ __launch_bounds__(256) void gcn_scatter_kernel(
    const int* __restrict__ src, const int* __restrict__ dst,
    const float* __restrict__ z, const float* __restrict__ thresh,
    const float* __restrict__ dinv, const float* __restrict__ m,
    float* __restrict__ agg, int E)
{
    int e = blockIdx.x * 8 + (threadIdx.x >> 5);
    if (e >= E) return;
    float zi = z[e];
    if (zi < *thresh) return;
    int s = src[e], d = dst[e];
    float nrm = dinv[s] * zi * dinv[d];
    int c4 = (threadIdx.x & 31) << 2;
    f4v mv = *(const f4v*)(m + s * 128 + c4);
#pragma unroll
    for (int j = 0; j < 4; ++j) atomicAdd(&agg[d * 128 + c4 + j], nrm * mv[j]);
}

__global__ void mul_kernel(float* a, const float* __restrict__ b, int n4) {
    int i = blockIdx.x * blockDim.x + threadIdx.x;
    if (i < n4) {
        f4v av = *(const f4v*)(a + i * 4);
        f4v bv = *(const f4v*)(b + i * 4);
#pragma unroll
        for (int j = 0; j < 4; ++j) av[j] *= bv[j];
        *(f4v*)(a + i * 4) = av;
    }
}

// h_new = (1-u)*tanh(nbuf) + u*prev
__global__ void gru_gate_kernel(const float* __restrict__ ubuf, const float* __restrict__ nbuf,
                                const float* __restrict__ prev, float* hnew, int n4) {
    int i = blockIdx.x * blockDim.x + threadIdx.x;
    if (i < n4) {
        f4v uv = *(const f4v*)(ubuf + i * 4);
        f4v nv = *(const f4v*)(nbuf + i * 4);
        f4v pv = *(const f4v*)(prev + i * 4);
        f4v r;
#pragma unroll
        for (int j = 0; j < 4; ++j) r[j] = (1.0f - uv[j]) * tanhf(nv[j]) + uv[j] * pv[j];
        *(f4v*)(hnew + i * 4) = r;
    }
}

// one wave per label edge: pred = sum_k hn[a][k]*hn[b][k]*wdec[k] + bdec
__global__ __launch_bounds__(256) void dec_kernel(
    const int* __restrict__ eli, const float* __restrict__ hn,
    const float* __restrict__ wdec, const float* __restrict__ bdec,
    float* pred, int L)
{
    int l = blockIdx.x * 8 + (threadIdx.x >> 5);
    if (l >= L) return;
    int a = eli[l], b = eli[L + l];
    int c4 = (threadIdx.x & 31) << 2;
    f4v av = *(const f4v*)(hn + a * 128 + c4);
    f4v bv = *(const f4v*)(hn + b * 128 + c4);
    f4v wv = *(const f4v*)(wdec + c4);
    float s = 0.0f;
#pragma unroll
    for (int j = 0; j < 4; ++j) s += av[j] * bv[j] * wv[j];
#pragma unroll
    for (int o = 16; o; o >>= 1) s += __shfl_xor(s, o, 32);
    if ((threadIdx.x & 31) == 0) pred[l] = s + bdec[0];
}

// ---------------------------------------------------------------------------
extern "C" void kernel_launch(void* const* d_in, const int* in_sizes, int n_in,
                              void* d_out, int out_size, void* d_ws, size_t ws_size,
                              hipStream_t stream) {
    const float* x       = (const float*)d_in[0];
    const int*   eidx    = (const int*)  d_in[1];
    const int*   eli     = (const int*)  d_in[2];
    const float* etime   = (const float*)d_in[3];
    const float* prev    = (const float*)d_in[4];
    const float* w_mlp   = (const float*)d_in[5];
    const float* b_mlp   = (const float*)d_in[6];
    const float* w_time  = (const float*)d_in[7];
    const float* b_time  = (const float*)d_in[8];
    const float* w_s1    = (const float*)d_in[9];
    const float* b_s1    = (const float*)d_in[10];
    const float* w_s2    = (const float*)d_in[11];
    const float* b_s2    = (const float*)d_in[12];
    const float* gcn_w   = (const float*)d_in[13];
    const float* gcn_b   = (const float*)d_in[14];
    const float* gru_wi  = (const float*)d_in[15];
    const float* gru_wh  = (const float*)d_in[16];
    const float* gru_bi  = (const float*)d_in[17];
    const float* gru_bh  = (const float*)d_in[18];
    const float* w_dec   = (const float*)d_in[19];
    const float* b_dec   = (const float*)d_in[20];

    const int H  = 128;
    const int E  = in_sizes[3];            // 400000
    const int N  = in_sizes[4] / H;        // 100000
    const int L  = in_sizes[2] / 2;        // 100000
    const int NH = N * H;
    const int NH4 = NH / 4;
    const int KTOP = (int)(0.8 * (double)E);

    const int* src = eidx;
    const int* dst = eidx + E;

    // workspace layout (floats), aligned chunks
    float*    ws    = (float*)d_ws;
    float*    z     = ws;                                   // E
    unsigned* hist  = (unsigned*)(ws + ((E + 63) & ~63));   // 4096
    unsigned* tminb = hist + 4096;                          // 1
    float*    thresh = (float*)(tminb + 1);                 // 1
    float*    deg  = (float*)(((uintptr_t)(thresh + 1) + 255) & ~(uintptr_t)255); // N (becomes dinv)
    float*    bufh = (float*)(((uintptr_t)(deg + N) + 255) & ~(uintptr_t)255);    // N*H
    float*    bufm = bufh + NH;                             // N*H
    float*    bufa = bufm + NH;                             // N*H
    float*    bufn = bufa + NH;                             // N*H

    float* pred = (float*)d_out;
    float* hnew = pred + L;

    const int TB = 256;
    const int gRows = (N + 63) / 64;
    const int gNH4  = (NH4 + TB - 1) / TB;
    const int gE    = (E + TB - 1) / TB;

    // 0) init hist / tmin / deg
    init_kernel<<<(max(N, 4096) + TB - 1) / TB, TB, 0, stream>>>(hist, tminb, deg, N);
    // 1) tmin reduction
    tmin_kernel<<<gE, TB, 0, stream>>>(etime, E, tminb);
    // 2) h = relu(x @ w_mlp + b_mlp)
    gemm128_kernel<<<gRows, TB, 0, stream>>>(x, N, 0, w_mlp, 128, 0, b_mlp, bufh, 0, 1);
    // 3) edge sampling scores z
    sample_kernel<<<(E + 63) / 64, TB, 0, stream>>>(prev, src, dst, etime, tminb,
                                                    w_time, b_time, w_s1, b_s1,
                                                    w_s2, b_s2, z, E);
    // 4) histogram top-k threshold
    hist_kernel<<<gE, TB, 0, stream>>>(z, E, hist);
    thresh_kernel<<<1, 32, 0, stream>>>(hist, thresh, KTOP);
    // 5) degrees -> dinv (in place)
    deg_kernel<<<gE, TB, 0, stream>>>(dst, z, thresh, deg, E);
    dinv_kernel<<<(N + TB - 1) / TB, TB, 0, stream>>>(deg, N);

    // 6) GCN layer 0: m = h@W0 ; agg = dinv^2*m + b0 ; scatter
    gemm128_kernel<<<gRows, TB, 0, stream>>>(bufh, N, 0, gcn_w, 128, 0, nullptr, bufm, 0, 0);
    agg_init_kernel<<<gNH4, TB, 0, stream>>>(bufm, deg, gcn_b, bufa, NH4);
    gcn_scatter_kernel<<<(E + 7) / 8, TB, 0, stream>>>(src, dst, z, thresh, deg, bufm, bufa, E);

    // 7) GCN layer 1: relu fused into A-staging of the GEMM; hfin = bufa
    gemm128_kernel<<<gRows, TB, 0, stream>>>(bufa, N, 1, gcn_w + 128 * 128, 128, 0, nullptr, bufm, 0, 0);
    agg_init_kernel<<<gNH4, TB, 0, stream>>>(bufm, deg, gcn_b + 128, bufa, NH4);
    gcn_scatter_kernel<<<(E + 7) / 8, TB, 0, stream>>>(src, dst, z, thresh, deg, bufm, bufa, E);
    float* hfin = bufa;

    // 8) GRU: r -> bufh, u -> bufm, n-input -> bufn
    gemm128_kernel<<<gRows, TB, 0, stream>>>(hfin, N, 0, gru_wi, 384, 0,   gru_bi,       bufh, 0, 0);
    gemm128_kernel<<<gRows, TB, 0, stream>>>(prev, N, 0, gru_wh, 384, 0,   gru_bh,       bufh, 1, 2); // r
    gemm128_kernel<<<gRows, TB, 0, stream>>>(hfin, N, 0, gru_wi, 384, 128, gru_bi + 128, bufm, 0, 0);
    gemm128_kernel<<<gRows, TB, 0, stream>>>(prev, N, 0, gru_wh, 384, 128, gru_bh + 128, bufm, 1, 2); // u
    gemm128_kernel<<<gRows, TB, 0, stream>>>(prev, N, 0, gru_wh, 384, 256, gru_bh + 256, bufn, 0, 0); // hn
    mul_kernel<<<gNH4, TB, 0, stream>>>(bufn, bufh, NH4);                                             // r*hn
    gemm128_kernel<<<gRows, TB, 0, stream>>>(hfin, N, 0, gru_wi, 384, 256, gru_bi + 256, bufn, 1, 0); // + inn
    gru_gate_kernel<<<gNH4, TB, 0, stream>>>(bufm, bufn, prev, hnew, NH4);

    // 9) decoder
    dec_kernel<<<(L + 7) / 8, TB, 0, stream>>>(eli, hnew, w_dec, b_dec, pred, L);
}
